// LatentMoETransition_3487513444976
// MI455X (gfx1250) — matrix operand order repaired and checked by reading, not compiled, not run on hardware
//
#include <hip/hip_runtime.h>
#include <hip/hip_bf16.h>

#define B_TOK     2048
#define DDIM      1024
#define NEXP      8
#define TOPK_K    2
#define MAX_STEPS 32
#define INNER     2048
#define LDT       72     // padded LDS row stride in bf16 elements (144B, 16B-aligned)
#define ABUF      (128 * LDT)   // elements per A LDS buffer
#define BBUF      (64  * LDT)   // elements per B LDS buffer

typedef __attribute__((ext_vector_type(16))) __bf16 v16bf;
typedef __attribute__((ext_vector_type(8)))  float  v8f;

__device__ __forceinline__ float gelu_exact(float x) {
  return 0.5f * x * (1.0f + erff(x * 0.70710678118654752f));
}

// Async DMA copy of 64B (4 x b128) global -> LDS, tracked by ASYNCcnt.
// Per CDNA5 ISA 10.x / 15.18: each lane copies from its global address + offset to its LDS address + offset.
__device__ __forceinline__ void async_copy_64B(unsigned lds_addr, const void* gaddr) {
  asm volatile(
      "global_load_async_to_lds_b128 %0, %1, off offset:0\n\t"
      "global_load_async_to_lds_b128 %0, %1, off offset:16\n\t"
      "global_load_async_to_lds_b128 %0, %1, off offset:32\n\t"
      "global_load_async_to_lds_b128 %0, %1, off offset:48"
      :: "v"(lds_addr), "v"(gaddr) : "memory");
}

__device__ __forceinline__ void wait_async0() {
  asm volatile("s_wait_asynccnt 0x0" ::: "memory");
}

// A fragment: 16x32 bf16. lane: M = lane&15; K-half kb = (lane>>4)*8.
// v0..3 hold K kb..kb+7, v4..7 hold K kb+16..kb+23 (per CDNA5 ISA 7.12.2).
__device__ __forceinline__ v16bf load_a_frag(const __bf16* base, int lane) {
  const int m  = lane & 15;
  const int kb = (lane >> 4) << 3;
  const __bf16* p = base + m * LDT + kb;
  union { uint4 u[2]; v16bf v; } t;
  t.u[0] = *(const uint4*)(p);        // K kb .. kb+7
  t.u[1] = *(const uint4*)(p + 16);   // K kb+16 .. kb+23
  return t.v;
}

// B fragment: 32x16 bf16. lane: N = lane&15; K-half kh = (lane>>4)*8; v0..3 K kh..kh+7, v4..7 K kh+16..kh+23.
__device__ __forceinline__ v16bf load_b_frag(const __bf16* base, int lane) {
  const int n  = lane & 15;
  const int kh = (lane >> 4) << 3;
  const __bf16* p = base + n * LDT + kh;
  union { uint4 u[2]; v16bf v; } t;
  t.u[0] = *(const uint4*)(p);        // K kh .. kh+7
  t.u[1] = *(const uint4*)(p + 16);   // K kh+16 .. kh+23
  return t.v;
}

// LDS-double-buffered GEMM core: C[128x64] += A[rows via rowmap, K](bf16) * W[N,K](fp32)^T
// 256 threads = 8 waves; wave w owns rows w*16..w*16+15, 4 N-subtiles of 16.
// K-tile = 64. A-tile k+1 staged by async DMA (global_load_async_to_lds_b128,
// ASYNCcnt path); W-tile staged through registers with fp32->bf16 conversion.
// Both overlap the wmma phase on buffer k. One barrier per K-tile.
__device__ __forceinline__ void gemm_core(
    const __bf16* __restrict__ A, const float* __restrict__ W,
    const int* rowmap, __bf16* sA, __bf16* sB,   // sA: 2*ABUF, sB: 2*BBUF
    int K, int n0, v8f acc[4])
{
  const int tid  = threadIdx.x;
  const int lane = tid & 31;
  const int wv   = tid >> 5;

  // A: thread stages 64B (4 x b128 async) of row (tid>>1), half (tid&1).
  const int arow = tid >> 1, ahalf = tid & 1;
  const __bf16* ga    = A + (size_t)rowmap[arow] * K + ahalf * 32;
  __bf16*       sa_st = sA + arow * LDT + ahalf * 32;
  // W: thread stages 64B (4 consecutive float4) of row (tid>>2), quarter (tid&3).
  const int wrow = tid >> 2, wq = tid & 3;
  const float* gw    = W + (size_t)(n0 + wrow) * K + wq * 16;
  __bf16*      sb_st = sB + wrow * LDT + wq * 16;

  // prologue: stage tile 0 into buffer 0
  async_copy_64B((unsigned)(unsigned long long)sa_st, (const void*)ga);
#pragma unroll
  for (int i = 0; i < 4; ++i) {
    const float4 f = *(const float4*)(gw + i * 4);
    union { __bf16 h[4]; uint2 u2; } pk;
    pk.h[0] = (__bf16)f.x; pk.h[1] = (__bf16)f.y;
    pk.h[2] = (__bf16)f.z; pk.h[3] = (__bf16)f.w;
    *(uint2*)(sb_st + i * 4) = pk.u2;
  }
  wait_async0();
  __syncthreads();

  int p = 0;
  for (int kk = 0; kk < K; kk += 64, p ^= 1) {
    // stage next tile into the other buffer (overlaps with wmma below)
    const int kn = kk + 64;
    if (kn < K) {
      async_copy_64B((unsigned)(unsigned long long)(sa_st + (p ^ 1) * ABUF),
                     (const void*)(ga + kn));
      __bf16* sb_n = sb_st + (p ^ 1) * BBUF;
#pragma unroll
      for (int i = 0; i < 4; ++i) {
        const float4 f = *(const float4*)(gw + kn + i * 4);
        union { __bf16 h[4]; uint2 u2; } pk;
        pk.h[0] = (__bf16)f.x; pk.h[1] = (__bf16)f.y;
        pk.h[2] = (__bf16)f.z; pk.h[3] = (__bf16)f.w;
        *(uint2*)(sb_n + i * 4) = pk.u2;
      }
    }

    // compute from current buffer: 2 K-steps of 32; per step 4 back-to-back wmma
    const __bf16* sa_c = sA + p * ABUF;
    const __bf16* sb_c = sB + p * BBUF;
#pragma unroll
    for (int s = 0; s < 2; ++s) {
      const v16bf a  = load_a_frag(sa_c + (wv << 4) * LDT + s * 32, lane);
      const v16bf b0 = load_b_frag(sb_c + 0  * LDT + s * 32, lane);
      const v16bf b1 = load_b_frag(sb_c + 16 * LDT + s * 32, lane);
      const v16bf b2 = load_b_frag(sb_c + 32 * LDT + s * 32, lane);
      const v16bf b3 = load_b_frag(sb_c + 48 * LDT + s * 32, lane);
      acc[0] = __builtin_amdgcn_wmma_f32_16x16x32_bf16(false, a, false, b0, (short)0, acc[0], false, false);
      acc[1] = __builtin_amdgcn_wmma_f32_16x16x32_bf16(false, a, false, b1, (short)0, acc[1], false, false);
      acc[2] = __builtin_amdgcn_wmma_f32_16x16x32_bf16(false, a, false, b2, (short)0, acc[2], false, false);
      acc[3] = __builtin_amdgcn_wmma_f32_16x16x32_bf16(false, a, false, b3, (short)0, acc[3], false, false);
    }
    wait_async0();   // async copies had the whole wmma phase to complete
    __syncthreads();
  }
}

// ---------------- prep: router + softmax + top2 + layernorm -> u(bf16) ----------------
__global__ __launch_bounds__(256)
void prep_kernel(const float* __restrict__ z, const int* __restrict__ step_id,
                 const float* __restrict__ ctx, const float* __restrict__ ln_w,
                 const float* __restrict__ ln_b, const float* __restrict__ stepe,
                 const float* __restrict__ rw, const float* __restrict__ rb,
                 __bf16* __restrict__ U, int* __restrict__ cnt,
                 int* __restrict__ pair_tok, float* __restrict__ pair_w,
                 float* __restrict__ prob_sum, float* __restrict__ ent_sum)
{
  const int lane  = threadIdx.x & 31;
  const int token = blockIdx.x * 8 + (threadIdx.x >> 5);
  int sid = step_id[0];
  sid = sid < 0 ? 0 : (sid > MAX_STEPS - 1 ? MAX_STEPS - 1 : sid);
  const float* zr = z     + (size_t)token * DDIM;
  const float* cr = ctx   + (size_t)token * DDIM;
  const float* er = stepe + (size_t)sid   * DDIM;

  float lg[NEXP] = {};
  float sz = 0.f, sz2 = 0.f;
  for (int i = 0; i < DDIM / 128; ++i) {
    const int d = lane * 4 + i * 128;
    const float4 z4 = *(const float4*)(zr + d);
    const float4 c4 = *(const float4*)(cr + d);
    const float4 e4 = *(const float4*)(er + d);
    const float sx = z4.x + c4.x, sy = z4.y + c4.y;
    const float s2 = z4.z + c4.z, sw = z4.w + c4.w;
    sz  += z4.x + z4.y + z4.z + z4.w;
    sz2 += z4.x*z4.x + z4.y*z4.y + z4.z*z4.z + z4.w*z4.w;
#pragma unroll
    for (int e = 0; e < NEXP; ++e) {
      const float* rwe = rw + (size_t)e * (2 * DDIM);
      const float4 w1 = *(const float4*)(rwe + d);
      const float4 w2 = *(const float4*)(rwe + DDIM + d);
      lg[e] += sx*w1.x + sy*w1.y + s2*w1.z + sw*w1.w
             + e4.x*w2.x + e4.y*w2.y + e4.z*w2.z + e4.w*w2.w;
    }
  }
#pragma unroll
  for (int off = 16; off > 0; off >>= 1) {
    sz  += __shfl_xor(sz,  off, 32);
    sz2 += __shfl_xor(sz2, off, 32);
#pragma unroll
    for (int e = 0; e < NEXP; ++e) lg[e] += __shfl_xor(lg[e], off, 32);
  }
  float p[NEXP];
  float mx = -3.4e38f;
#pragma unroll
  for (int e = 0; e < NEXP; ++e) { lg[e] += rb[e]; mx = fmaxf(mx, lg[e]); }
  float den = 0.f;
#pragma unroll
  for (int e = 0; e < NEXP; ++e) { p[e] = expf(lg[e] - mx); den += p[e]; }
  const float rden = 1.f / den;
  float ent = 0.f;
#pragma unroll
  for (int e = 0; e < NEXP; ++e) { p[e] *= rden; ent -= p[e] * logf(fmaxf(p[e], 1e-8f)); }
  int e0 = 0; float p0 = p[0];
#pragma unroll
  for (int e = 1; e < NEXP; ++e) if (p[e] > p0) { p0 = p[e]; e0 = e; }
  int e1 = 0; float p1 = -1.f;
#pragma unroll
  for (int e = 0; e < NEXP; ++e) if (e != e0 && p[e] > p1) { p1 = p[e]; e1 = e; }

  if (lane == 0) {
    atomicAdd(ent_sum, ent);
#pragma unroll
    for (int e = 0; e < NEXP; ++e) atomicAdd(&prob_sum[e], p[e]);
    int pos0 = atomicAdd(&cnt[e0], 1);
    pair_tok[e0 * B_TOK + pos0] = token; pair_w[e0 * B_TOK + pos0] = p0;
    int pos1 = atomicAdd(&cnt[e1], 1);
    pair_tok[e1 * B_TOK + pos1] = token; pair_w[e1 * B_TOK + pos1] = p1;
  }

  const float mu   = sz * (1.0f / DDIM);
  const float var  = sz2 * (1.0f / DDIM) - mu * mu;
  const float rstd = rsqrtf(var + 1e-5f);
  for (int i = 0; i < DDIM / 128; ++i) {
    const int d = lane * 4 + i * 128;
    const float4 z4 = *(const float4*)(zr + d);
    const float4 w4 = *(const float4*)(ln_w + d);
    const float4 b4 = *(const float4*)(ln_b + d);
    union { __bf16 h[4]; uint2 u2; } pk;
    pk.h[0] = (__bf16)((z4.x - mu) * rstd * w4.x + b4.x);
    pk.h[1] = (__bf16)((z4.y - mu) * rstd * w4.y + b4.y);
    pk.h[2] = (__bf16)((z4.z - mu) * rstd * w4.z + b4.z);
    pk.h[3] = (__bf16)((z4.w - mu) * rstd * w4.w + b4.w);
    *(uint2*)(U + (size_t)token * DDIM + d) = pk.u2;
  }
}

__global__ void init_kernel(int* cnt, float* prob_sum, float* ent_sum, int* offs) {
  const int t = threadIdx.x;
  if (t < NEXP) { cnt[t] = 0; prob_sum[t] = 0.f; offs[t] = 0; }
  if (t == 0) ent_sum[0] = 0.f;
}

__global__ void offsets_kernel(const int* cnt, int* offs) {
  if (threadIdx.x == 0 && blockIdx.x == 0) {
    int a = 0;
    for (int e = 0; e < NEXP; ++e) { offs[e] = a; a += cnt[e]; }
  }
}

// ---------------- shared expert fc1: H_s = gelu(U @ W^T + b), bf16 out ----------------
__global__ __launch_bounds__(256, 1)
void shared_fc1_kernel(const __bf16* __restrict__ U, const float* __restrict__ W,
                       const float* __restrict__ bias, __bf16* __restrict__ H)
{
  __shared__ __bf16 sA[2 * ABUF];
  __shared__ __bf16 sB[2 * BBUF];
  __shared__ int rowmap[128];
  const int n0 = blockIdx.x * 64;
  const int m0 = blockIdx.y * 128;
  const int tid = threadIdx.x;
  if (tid < 128) rowmap[tid] = m0 + tid;
  __syncthreads();
  v8f acc[4] = {};
  gemm_core(U, W, rowmap, sA, sB, DDIM, n0, acc);
  const int lane = tid & 31, wv = tid >> 5;
  const int mb = m0 + (wv << 4) + ((lane >> 4) << 3);
  const int nb = n0 + (lane & 15);
#pragma unroll
  for (int j = 0; j < 4; ++j)
#pragma unroll
    for (int r = 0; r < 8; ++r) {
      const int col = nb + j * 16;
      const float x = acc[j][r] + bias[col];
      H[(size_t)(mb + r) * INNER + col] = (__bf16)gelu_exact(x);
    }
}

// ---------------- shared expert fc2 + residual: out = z + H_s @ W^T + b ----------------
__global__ __launch_bounds__(256, 1)
void shared_fc2_kernel(const __bf16* __restrict__ H, const float* __restrict__ W,
                       const float* __restrict__ bias, const float* __restrict__ z,
                       float* __restrict__ out)
{
  __shared__ __bf16 sA[2 * ABUF];
  __shared__ __bf16 sB[2 * BBUF];
  __shared__ int rowmap[128];
  const int n0 = blockIdx.x * 64;
  const int m0 = blockIdx.y * 128;
  const int tid = threadIdx.x;
  if (tid < 128) rowmap[tid] = m0 + tid;
  __syncthreads();
  v8f acc[4] = {};
  gemm_core(H, W, rowmap, sA, sB, INNER, n0, acc);
  const int lane = tid & 31, wv = tid >> 5;
  const int mb = m0 + (wv << 4) + ((lane >> 4) << 3);
  const int nb = n0 + (lane & 15);
#pragma unroll
  for (int j = 0; j < 4; ++j)
#pragma unroll
    for (int r = 0; r < 8; ++r) {
      const int col = nb + j * 16;
      const size_t idx = (size_t)(mb + r) * DDIM + col;
      out[idx] = z[idx] + acc[j][r] + bias[col];
    }
}

// ---------------- routed expert fc1 (gathered rows) ----------------
__global__ __launch_bounds__(256, 1)
void expert_fc1_kernel(const __bf16* __restrict__ U, const float* __restrict__ W_all,
                       const float* __restrict__ bias_all,
                       const int* __restrict__ cnt, const int* __restrict__ offs,
                       const int* __restrict__ pair_tok, __bf16* __restrict__ H)
{
  const int e = blockIdx.z;
  const int cnt_e = cnt[e];
  const int m0 = blockIdx.y * 128;
  if (m0 >= cnt_e) return;
  const int off_e = offs[e];
  __shared__ __bf16 sA[2 * ABUF];
  __shared__ __bf16 sB[2 * BBUF];
  __shared__ int rowmap[128];
  const int n0 = blockIdx.x * 64;
  const int tid = threadIdx.x;
  if (tid < 128) {
    int pos = m0 + tid; pos = pos < cnt_e ? pos : cnt_e - 1;
    rowmap[tid] = pair_tok[e * B_TOK + pos];
  }
  __syncthreads();
  v8f acc[4] = {};
  gemm_core(U, W_all + (size_t)e * INNER * DDIM, rowmap, sA, sB, DDIM, n0, acc);
  const int lane = tid & 31, wv = tid >> 5;
  const int pb = m0 + (wv << 4) + ((lane >> 4) << 3);
  const int nb = n0 + (lane & 15);
#pragma unroll
  for (int j = 0; j < 4; ++j)
#pragma unroll
    for (int r = 0; r < 8; ++r) {
      const int pos = pb + r;
      if (pos < cnt_e) {
        const int col = nb + j * 16;
        const float x = acc[j][r] + bias_all[e * INNER + col];
        H[(size_t)(off_e + pos) * INNER + col] = (__bf16)gelu_exact(x);
      }
    }
}

// ---------------- routed expert fc2: scatter-add prob * (H_e @ W^T + b) ----------------
__global__ __launch_bounds__(256, 1)
void expert_fc2_kernel(const __bf16* __restrict__ H, const float* __restrict__ W_all,
                       const float* __restrict__ bias_all,
                       const int* __restrict__ cnt, const int* __restrict__ offs,
                       const int* __restrict__ pair_tok, const float* __restrict__ pair_w,
                       float* __restrict__ out)
{
  const int e = blockIdx.z;
  const int cnt_e = cnt[e];
  const int m0 = blockIdx.y * 128;
  if (m0 >= cnt_e) return;
  const int off_e = offs[e];
  __shared__ __bf16 sA[2 * ABUF];
  __shared__ __bf16 sB[2 * BBUF];
  __shared__ int rowmap[128];
  const int n0 = blockIdx.x * 64;
  const int tid = threadIdx.x;
  if (tid < 128) {
    int pos = m0 + tid; pos = pos < cnt_e ? pos : cnt_e - 1;
    rowmap[tid] = off_e + pos;
  }
  __syncthreads();
  v8f acc[4] = {};
  gemm_core(H, W_all + (size_t)e * DDIM * INNER, rowmap, sA, sB, INNER, n0, acc);
  const int lane = tid & 31, wv = tid >> 5;
  const int pb = m0 + (wv << 4) + ((lane >> 4) << 3);
  const int nb = n0 + (lane & 15);
#pragma unroll
  for (int j = 0; j < 4; ++j)
#pragma unroll
    for (int r = 0; r < 8; ++r) {
      const int pos = pb + r;
      if (pos < cnt_e) {
        const int tok = pair_tok[e * B_TOK + pos];
        const float w = pair_w[e * B_TOK + pos];
        const int col = nb + j * 16;
        atomicAdd(&out[(size_t)tok * DDIM + col],
                  w * (acc[j][r] + bias_all[e * DDIM + col]));
      }
    }
}

__global__ void finalize_kernel(const float* prob_sum, const float* ent_sum,
                                float* out_tail) {
  if (threadIdx.x == 0 && blockIdx.x == 0) {
    float bl = 0.f;
    for (int e = 0; e < NEXP; ++e) {
      const float d = prob_sum[e] * (1.0f / B_TOK) - 1.0f / NEXP;
      bl += d * d;
    }
    out_tail[0] = bl * (1.0f / NEXP);
    out_tail[1] = ent_sum[0] * (1.0f / B_TOK);
  }
}

extern "C" void kernel_launch(void* const* d_in, const int* in_sizes, int n_in,
                              void* d_out, int out_size, void* d_ws, size_t ws_size,
                              hipStream_t stream) {
  const float* z_t    = (const float*)d_in[0];
  const int*   step   = (const int*)  d_in[1];
  const float* ctx    = (const float*)d_in[2];
  const float* ln_w   = (const float*)d_in[3];
  const float* ln_b   = (const float*)d_in[4];
  const float* se     = (const float*)d_in[5];
  const float* rw     = (const float*)d_in[6];
  const float* rb     = (const float*)d_in[7];
  const float* fc1_w  = (const float*)d_in[8];
  const float* fc1_b  = (const float*)d_in[9];
  const float* fc2_w  = (const float*)d_in[10];
  const float* fc2_b  = (const float*)d_in[11];
  const float* sfc1_w = (const float*)d_in[12];
  const float* sfc1_b = (const float*)d_in[13];
  const float* sfc2_w = (const float*)d_in[14];
  const float* sfc2_b = (const float*)d_in[15];
  float* out = (float*)d_out;

  char* ws = (char*)d_ws;
  __bf16* U  = (__bf16*)ws;  ws += (size_t)B_TOK * DDIM * 2;
  __bf16* Hs = (__bf16*)ws;  ws += (size_t)B_TOK * INNER * 2;
  __bf16* He = (__bf16*)ws;  ws += (size_t)B_TOK * TOPK_K * INNER * 2;
  int*   pair_tok = (int*)ws;   ws += (size_t)NEXP * B_TOK * 4;
  float* pair_w   = (float*)ws; ws += (size_t)NEXP * B_TOK * 4;
  int*   cnt      = (int*)ws;   ws += 64;
  int*   offs     = (int*)ws;   ws += 64;
  float* prob_sum = (float*)ws; ws += 64;
  float* ent_sum  = (float*)ws; ws += 64;

  init_kernel<<<1, 32, 0, stream>>>(cnt, prob_sum, ent_sum, offs);
  prep_kernel<<<B_TOK / 8, 256, 0, stream>>>(z_t, step, ctx, ln_w, ln_b, se, rw, rb,
                                             U, cnt, pair_tok, pair_w, prob_sum, ent_sum);
  offsets_kernel<<<1, 32, 0, stream>>>(cnt, offs);
  shared_fc1_kernel<<<dim3(INNER / 64, B_TOK / 128), 256, 0, stream>>>(U, sfc1_w, sfc1_b, Hs);
  expert_fc1_kernel<<<dim3(INNER / 64, B_TOK / 128, NEXP), 256, 0, stream>>>(
      U, fc1_w, fc1_b, cnt, offs, pair_tok, He);
  shared_fc2_kernel<<<dim3(DDIM / 64, B_TOK / 128), 256, 0, stream>>>(Hs, sfc2_w, sfc2_b, z_t, out);
  expert_fc2_kernel<<<dim3(DDIM / 64, B_TOK / 128, NEXP), 256, 0, stream>>>(
      He, fc2_w, fc2_b, cnt, offs, pair_tok, pair_w, out);
  finalize_kernel<<<1, 32, 0, stream>>>(prob_sum, ent_sum, out + (size_t)B_TOK * DDIM);
}
